// MultiScaleDeformableAttention_61984968015933
// MI455X (gfx1250) — matrix operand order repaired and checked by reading, not compiled, not run on hardware
//
#include <hip/hip_runtime.h>
#include <hip/hip_bf16.h>

// ---------------------------------------------------------------------------
// Multi-Scale Deformable Attention, MI455X (gfx1250, wave32, WMMA)
//
// Constants baked from the reference:
//   E=256, NH=8, NL=4, NP=4, D=32, B=2, L=17821
//   levels: (100,134) (50,67) (25,34) (13,17); starts 0,13400,16750,17600
// ---------------------------------------------------------------------------

typedef __attribute__((ext_vector_type(16))) _Float16 v16h;
typedef __attribute__((ext_vector_type(8)))  float    v8f;

#define MSDA_E   256
#define MSDA_NH  8
#define MSDA_NL  4
#define MSDA_NP  4
#define MSDA_D   32
#define MSDA_B   2
#define MSDA_L   17821
#define MSDA_M   (MSDA_B * MSDA_L)   // 35642 GEMM rows

__device__ __constant__ int   c_lvlH[4]     = {100, 50, 25, 13};
__device__ __constant__ int   c_lvlW[4]     = {134, 67, 34, 17};
__device__ __constant__ int   c_lvlStart[4] = {0, 13400, 16750, 17600};
__device__ __constant__ float c_lvlHf[4]    = {100.f, 50.f, 25.f, 13.f};
__device__ __constant__ float c_lvlWf[4]    = {134.f, 67.f, 34.f, 17.f};

// ---------------------------------------------------------------------------
// Weight prep: fp32 row-major (K x N)  ->  f16 transposed (N x K).
// Puts the B operand in the per-lane-contiguous-K form this code uses for the WMMA
// 32x16 f16 B operand (lane = column, 16 contiguous K per half-wave).
// ---------------------------------------------------------------------------
__global__ void msda_prep_weights(const float* __restrict__ W,
                                  _Float16* __restrict__ Wt,
                                  int K, int N) {
    int idx = blockIdx.x * blockDim.x + threadIdx.x;
    if (idx >= K * N) return;
    int n = idx / K;
    int k = idx - n * K;
    Wt[idx] = (_Float16)W[k * N + n];
}

// ---------------------------------------------------------------------------
// A-stationary K=256 GEMM: C[M x N] = A[M x 256] * Wt[N x 256]^T + bias
//   in_lbe:  A rows are (L,B,E) interleaved (row m -> src[i,b,:], m=b*L+i)
//   out_lbe: C rows stored (L,B,N) interleaved (final output transpose)
//
// Each wave loads+converts its 16x256 A slab ONCE (8 x v16h fragments held
// in registers), then sweeps `npern` output columns in 32-wide passes of
// 8 x v_wmma_f32_16x16x32_f16. Workgroup = 8 waves (4 M x 2 N subtiles)
// -> 64-row block, npern columns per block.
// ---------------------------------------------------------------------------
__global__ __launch_bounds__(256)
void msda_gemm256(const float*    __restrict__ A,
                  const _Float16* __restrict__ Wt,
                  const float*    __restrict__ bias,
                  float*          __restrict__ C,
                  int N, int npern, int in_lbe, int out_lbe) {
    const int lane  = threadIdx.x & 31;
    const int wave  = threadIdx.x >> 5;
    const int wm    = wave & 3;
    const int wn    = wave >> 2;
    const int m0    = blockIdx.x * 64 + wm * 16;
    const int nbase = blockIdx.y * npern;
    const int half  = lane >> 4;      // K-half selector for A/B fragments
    const int l16   = lane & 15;

    // Warm L2 for this lane's first transposed-weight row (global_prefetch_b8)
    __builtin_prefetch(Wt + (size_t)(nbase + wn * 16 + l16) * 256, 0, 3);

    // A row this lane feeds (clamped; stores predicated later so EXEC stays
    // all-ones through every WMMA as the ISA requires).
    int m = m0 + l16;
    if (m >= MSDA_M) m = MSDA_M - 1;
    size_t abase;
    if (in_lbe) {
        int b = m / MSDA_L;
        int i = m - b * MSDA_L;
        abase = ((size_t)i * MSDA_B + b) * MSDA_E;
    } else {
        abase = (size_t)m * MSDA_E;
    }

    // Load + convert the full 16x256 A slab once (A-stationary).
    // A layout per 32-K block (matches B): half 0 -> K 0..15,
    // half 1 -> K 16..31  => one contiguous 16-float run per lane.
    v16h afrag[8];
#pragma unroll
    for (int kb8 = 0; kb8 < 8; ++kb8) {
        const int kb = kb8 * 32;
        const float4* pa0 = reinterpret_cast<const float4*>(A + abase + kb + 16 * half);
        const float4* pa1 = reinterpret_cast<const float4*>(A + abase + kb + 8 + 16 * half);
        float4 a0 = pa0[0], a1 = pa0[1];
        float4 a2 = pa1[0], a3 = pa1[1];
        v16h av;
        av[0]  = (_Float16)a0.x; av[1]  = (_Float16)a0.y;
        av[2]  = (_Float16)a0.z; av[3]  = (_Float16)a0.w;
        av[4]  = (_Float16)a1.x; av[5]  = (_Float16)a1.y;
        av[6]  = (_Float16)a1.z; av[7]  = (_Float16)a1.w;
        av[8]  = (_Float16)a2.x; av[9]  = (_Float16)a2.y;
        av[10] = (_Float16)a2.z; av[11] = (_Float16)a2.w;
        av[12] = (_Float16)a3.x; av[13] = (_Float16)a3.y;
        av[14] = (_Float16)a3.z; av[15] = (_Float16)a3.w;
        afrag[kb8] = av;
    }

    const int passes = npern >> 5;   // 32 columns per pass (2 N-waves x 16)
    for (int np = 0; np < passes; ++np) {
        const int n = nbase + np * 32 + wn * 16 + l16;   // grid keeps n < N
        const _Float16* wrow = Wt + (size_t)n * 256 + 16 * half;

        v8f acc = {};
#pragma unroll
        for (int kb8 = 0; kb8 < 8; ++kb8) {
            // B fragment: 16 contiguous f16 of the transposed weight row
            // (32B aligned: n*256 elems + kb + 16*half)
            v16h bv = *reinterpret_cast<const v16h*>(wrow + kb8 * 32);
            acc = __builtin_amdgcn_wmma_f32_16x16x32_f16(
                /*neg_a=*/false, afrag[kb8], /*neg_b=*/false, bv,
                /*c_mod=*/(short)0, acc, /*reuse_a=*/false, /*reuse_b=*/false);
        }

        const float bn = bias[n];
#pragma unroll
        for (int r = 0; r < 8; ++r) {
            int mo = m0 + r + 8 * half;       // C layout: VGPR r -> M = r + 8*half
            if (mo < MSDA_M) {
                size_t oidx;
                if (out_lbe) {
                    int b = mo / MSDA_L;
                    int i = mo - b * MSDA_L;
                    oidx = ((size_t)i * MSDA_B + b) * N + n;
                } else {
                    oidx = (size_t)mo * N + n;
                }
                C[oidx] = acc[r] + bn;
            }
        }
    }
}

// ---------------------------------------------------------------------------
// Per (b,q,h,l): softmax over the NP=4 attention logits (in place) and
// convert raw offsets -> normalized sampling locations (in place).
// ---------------------------------------------------------------------------
__global__ void msda_loc_softmax(float* __restrict__ off,    // (B,Lq,NH,NL,NP,2)
                                 float* __restrict__ attn,   // (B,Lq,NH,NL,NP)
                                 const float* __restrict__ ref) { // (B,Lq,NL,2)
    size_t t = (size_t)blockIdx.x * blockDim.x + threadIdx.x;
    const size_t total = (size_t)MSDA_B * MSDA_L * MSDA_NH * MSDA_NL;
    if (t >= total) return;
    int    l  = (int)(t & 3);
    int    h  = (int)((t >> 2) & 7);
    size_t bq = t >> 5;                        // b*Lq + q

    float* ap = attn + ((bq * MSDA_NH + h) * MSDA_NL + l) * MSDA_NP;
    float a0 = ap[0], a1 = ap[1], a2 = ap[2], a3 = ap[3];
    float mx = fmaxf(fmaxf(a0, a1), fmaxf(a2, a3));
    float e0 = __expf(a0 - mx), e1 = __expf(a1 - mx);
    float e2 = __expf(a2 - mx), e3 = __expf(a3 - mx);
    float inv = 1.0f / (e0 + e1 + e2 + e3);
    ap[0] = e0 * inv; ap[1] = e1 * inv; ap[2] = e2 * inv; ap[3] = e3 * inv;

    float rx = ref[(bq * MSDA_NL + l) * 2 + 0];
    float ry = ref[(bq * MSDA_NL + l) * 2 + 1];
    float invW = 1.0f / c_lvlWf[l];
    float invH = 1.0f / c_lvlHf[l];
    float* op = off + ((bq * MSDA_NH + h) * MSDA_NL + l) * (MSDA_NP * 2);
#pragma unroll
    for (int p = 0; p < MSDA_NP; ++p) {
        float ox = op[2 * p], oy = op[2 * p + 1];
        op[2 * p]     = rx + ox * invW;
        op[2 * p + 1] = ry + oy * invH;
    }
}

// ---------------------------------------------------------------------------
// Bilinear gather + weighted sum. One wave per (b,q,h); lane = channel
// (D = 32 = wave32, so every corner gather is one coalesced 128B load).
// ---------------------------------------------------------------------------
__global__ __launch_bounds__(256)
void msda_sample(const float* __restrict__ vproj,  // (B,Lin,NH,D)
                 const float* __restrict__ loc,    // (B,Lq,NH,NL,NP,2)
                 const float* __restrict__ aw,     // (B,Lq,NH,NL,NP) softmaxed
                 float* __restrict__ out) {        // (B,Lq,E)
    int gw   = blockIdx.x * (blockDim.x >> 5) + (threadIdx.x >> 5);
    int lane = threadIdx.x & 31;
    const int totalw = MSDA_B * MSDA_L * MSDA_NH;
    if (gw >= totalw) return;
    int    h  = gw & 7;
    size_t bq = (size_t)(gw >> 3);             // b*Lq + q
    int    b  = (int)(bq / MSDA_L);

    const float* lp = loc + (bq * MSDA_NH + h) * (MSDA_NL * MSDA_NP * 2);
    const float* wp = aw  + (bq * MSDA_NH + h) * (MSDA_NL * MSDA_NP);

    float acc = 0.0f;
#pragma unroll
    for (int l = 0; l < MSDA_NL; ++l) {
        const int Hh = c_lvlH[l], Ww = c_lvlW[l];
        const float* vbase = vproj
            + (((size_t)b * MSDA_L + c_lvlStart[l]) * MSDA_NH + h) * MSDA_D + lane;
#pragma unroll
        for (int p = 0; p < MSDA_NP; ++p) {
            float x = lp[(l * MSDA_NP + p) * 2 + 0] * (float)Ww - 0.5f;
            float y = lp[(l * MSDA_NP + p) * 2 + 1] * (float)Hh - 0.5f;
            float x0f = floorf(x), y0f = floorf(y);
            float dx = x - x0f, dy = y - y0f;
            int xi = (int)x0f, yi = (int)y0f;
            float wgt = wp[l * MSDA_NP + p];
            float w00 = (1.0f - dx) * (1.0f - dy) * wgt;
            float w10 = dx * (1.0f - dy) * wgt;
            float w01 = (1.0f - dx) * dy * wgt;
            float w11 = dx * dy * wgt;
            // stride between spatial cells in vproj = NH*D = 256 floats
            if (yi >= 0 && yi < Hh && xi >= 0 && xi < Ww)
                acc += w00 * vbase[(size_t)(yi * Ww + xi) * 256];
            if (yi >= 0 && yi < Hh && xi + 1 >= 0 && xi + 1 < Ww)
                acc += w10 * vbase[(size_t)(yi * Ww + xi + 1) * 256];
            if (yi + 1 >= 0 && yi + 1 < Hh && xi >= 0 && xi < Ww)
                acc += w01 * vbase[(size_t)((yi + 1) * Ww + xi) * 256];
            if (yi + 1 >= 0 && yi + 1 < Hh && xi + 1 >= 0 && xi + 1 < Ww)
                acc += w11 * vbase[(size_t)((yi + 1) * Ww + xi + 1) * 256];
        }
    }
    out[bq * MSDA_E + h * MSDA_D + lane] = acc;
}

// ---------------------------------------------------------------------------
// Host-side orchestration (graph-capture safe: launches only, all on stream).
// ---------------------------------------------------------------------------
extern "C" void kernel_launch(void* const* d_in, const int* in_sizes, int n_in,
                              void* d_out, int out_size, void* d_ws, size_t ws_size,
                              hipStream_t stream) {
    const float* query    = (const float*)d_in[0];   // (L,B,E)
    const float* value    = (const float*)d_in[1];   // (L,B,E)
    const float* refpts   = (const float*)d_in[2];   // (B,L,NL,2)
    // d_in[3] spatial_shapes (int64) — compile-time constants, unused
    const float* W_value  = (const float*)d_in[4];
    const float* b_value  = (const float*)d_in[5];
    const float* W_offset = (const float*)d_in[6];
    const float* b_offset = (const float*)d_in[7];
    const float* W_attn   = (const float*)d_in[8];
    const float* b_attn   = (const float*)d_in[9];
    const float* W_out    = (const float*)d_in[10];
    const float* b_out    = (const float*)d_in[11];
    float* out = (float*)d_out;                      // (L,B,E)

    // Workspace layout (all 256B aligned)
    char* ws = (char*)d_ws;
    const size_t sz_ble256 = (size_t)MSDA_M * 256 * sizeof(float);  // 36.5 MB
    const size_t sz_ble128 = (size_t)MSDA_M * 128 * sizeof(float);  // 18.2 MB
    float* vproj   = (float*)(ws);                                   // (B,Lin,NH,D)
    float* attnout = (float*)(ws + sz_ble256);                       // (B,Lq,E)
    float* offbuf  = (float*)(ws + 2 * sz_ble256);                   // raw offsets -> loc
    float* attnbuf = (float*)(ws + 3 * sz_ble256);                   // raw attn -> softmax
    char*  wbase   = ws + 3 * sz_ble256 + sz_ble128;
    _Float16* wt_value  = (_Float16*)(wbase);
    _Float16* wt_offset = (_Float16*)(wbase + 256 * 256 * 2);
    _Float16* wt_attn   = (_Float16*)(wbase + 2 * (256 * 256 * 2));
    _Float16* wt_out    = (_Float16*)(wbase + 2 * (256 * 256 * 2) + 256 * 128 * 2);

    // 1) Weights -> f16, transposed to N-major (B-operand native layout)
    msda_prep_weights<<<(256 * 256 + 255) / 256, 256, 0, stream>>>(W_value,  wt_value,  256, 256);
    msda_prep_weights<<<(256 * 256 + 255) / 256, 256, 0, stream>>>(W_offset, wt_offset, 256, 256);
    msda_prep_weights<<<(256 * 128 + 255) / 256, 256, 0, stream>>>(W_attn,   wt_attn,   256, 128);
    msda_prep_weights<<<(256 * 256 + 255) / 256, 256, 0, stream>>>(W_out,    wt_out,    256, 256);

    const int mblocks = (MSDA_M + 63) / 64;

    // 2) Value projection: (B,Lin,E) = value(L,B,E) @ W_value
    //    N=256 split into 2 y-blocks of 128 cols (4 passes each, A-stationary)
    msda_gemm256<<<dim3(mblocks, 2), 256, 0, stream>>>(
        value, wt_value, b_value, vproj, 256, 128, /*in_lbe=*/1, /*out_lbe=*/0);

    // 3) Offset + attention projections from query
    msda_gemm256<<<dim3(mblocks, 2), 256, 0, stream>>>(
        query, wt_offset, b_offset, offbuf, 256, 128, 1, 0);
    msda_gemm256<<<dim3(mblocks, 1), 256, 0, stream>>>(
        query, wt_attn, b_attn, attnbuf, 128, 128, 1, 0);

    // 4) Softmax over points + sampling-location computation (in place)
    {
        size_t total = (size_t)MSDA_B * MSDA_L * MSDA_NH * MSDA_NL;
        msda_loc_softmax<<<(unsigned)((total + 255) / 256), 256, 0, stream>>>(
            offbuf, attnbuf, refpts);
    }

    // 5) Bilinear sampling: one wave per (b,q,h), lane = channel
    {
        int totalw = MSDA_B * MSDA_L * MSDA_NH;
        msda_sample<<<(totalw + 7) / 8, 256, 0, stream>>>(vproj, offbuf, attnbuf, attnout);
    }

    // 6) Output projection with (B,Lq,E) -> (Lq,B,E) transpose on store
    msda_gemm256<<<dim3(mblocks, 2), 256, 0, stream>>>(
        attnout, wt_out, b_out, out, 256, 128, /*in_lbe=*/0, /*out_lbe=*/1);
}
